// Seq2SeqModel_64854006170143
// MI455X (gfx1250) — matrix operand order repaired and checked by reading, not compiled, not run on hardware
//
#include <hip/hip_runtime.h>
#include <hip/hip_bf16.h>

// ---------------------------------------------------------------------------
// Seq2Seq LSTM (encoder 128 steps + decoder 127 steps + FC) for gfx1250.
// Core compute: v_wmma_f32_16x16x32_f16 on f16 h/Whh with fp32 accumulation.
// Whh tiles staged through LDS with async global->LDS copies (double-buffered,
// 64-wide K slices, one barrier per slice). Input projection precomputed as a
// (V,4H) table (emb @ Wih.T + biases) -> the x-path is a gather, not a GEMM.
// ---------------------------------------------------------------------------

typedef __attribute__((ext_vector_type(16))) _Float16     v16h;
typedef __attribute__((ext_vector_type(8)))  float        v8f;
typedef __attribute__((ext_vector_type(4)))  unsigned int u32x4;
typedef int i32x4 __attribute__((vector_size(4 * sizeof(int))));

#define AS1 __attribute__((address_space(1)))
#define AS3 __attribute__((address_space(3)))

#define E_  256
#define H_  512
#define G_  2048   // 4*H
#define VI_ 128
#define VO_ 128
#define B_  1024
#define S_  128
#define T_  128

#define KSLICE  64   // K width staged per barrier (2 WMMA K-steps)
#define BSTRIDE 72   // LDS halves per B column (64 data + 8 pad; 144B, 16B-aligned,
                     // 36-dword lane stride -> 16 distinct banks of 64)

#if __has_builtin(__builtin_amdgcn_global_load_async_to_lds_b128)
#define HAS_ASYNC_LDS 1
#else
#define HAS_ASYNC_LDS 0
#endif

__device__ __forceinline__ float sigf(float x) { return 1.0f / (1.0f + __expf(-x)); }

// A fragment (16x32 f16, M x K) from row-major f16 (lda = H_):
// lane<16:  M=m0+lane,    K = {k0..k0+7,  k0+16..k0+23}
// lane>=16: M=m0+lane-16, K = {k0+8..k0+15, k0+24..k0+31}
__device__ __forceinline__ v16h load_a_frag(const _Float16* __restrict__ p) {
    union { v16h v; struct { u32x4 lo; u32x4 hi; } q; } u;
    u.q.lo = *reinterpret_cast<const u32x4*>(p);        // 8 halves
    u.q.hi = *reinterpret_cast<const u32x4*>(p + 16);   // 8 halves, +32B
    return u.v;
}

// B fragment: lane<16 -> K {0..7, 16..23}, lane>=16 -> K {8..15, 24..31} (p pre-offset by 8*lh).
// 16B-aligned loads only (source may be 16B aligned, not 32B).
__device__ __forceinline__ v16h load_b_frag16(const _Float16* p) {
    union { v16h v; struct { u32x4 lo; u32x4 hi; } q; } u;
    u.q.lo = *reinterpret_cast<const u32x4*>(p);
    u.q.hi = *reinterpret_cast<const u32x4*>(p + 16);
    return u.v;
}

// Cooperatively stage one K-slice (KSLICE k) of the 4 gate B-tiles (32 n each)
// into LDS: 1024 chunks of 16B, 4 per thread. LDS layout: [q][n][k] with
// per-column stride BSTRIDE halves (bank-conflict-free fragment reads).
__device__ __forceinline__ void stage_b(const _Float16* __restrict__ whh,
                                        int n0b, int kk,
                                        _Float16* lbuf, int tid)
{
#pragma unroll
    for (int it = 0; it < 4; ++it) {
        const int c  = tid + it * 256;      // 0..1023
        const int q  = c >> 8;              // gate 0..3
        const int n  = (c >> 3) & 31;       // column in block tile
        const int k8 = c & 7;               // 8-half chunk in K slice
        const _Float16* g = whh + (size_t)(q * H_ + n0b + n) * H_ + kk + k8 * 8;
        _Float16*       l = lbuf + (q * 32 + n) * BSTRIDE + k8 * 8;
#if HAS_ASYNC_LDS
        __builtin_amdgcn_global_load_async_to_lds_b128(
            (AS1 i32x4*)(AS1 void*)g, (AS3 i32x4*)(AS3 void*)l, 0, 0);
#else
        *reinterpret_cast<u32x4*>(l) = *reinterpret_cast<const u32x4*>(g);
#endif
    }
}

__device__ __forceinline__ void wait_async_lds() {
#if HAS_ASYNC_LDS
#if __has_builtin(__builtin_amdgcn_s_wait_asynccnt)
    __builtin_amdgcn_s_wait_asynccnt(0);
#else
    asm volatile("s_wait_asynccnt 0x0" ::: "memory");
#endif
#endif
}

// ------------------- fused LSTM cell: gates = h@Whh.T + xtab[tok] ----------
// block = 256 threads = 8 waves laid out 4(M) x 2(N); block tile 64 x 32.
// grid = (B/64, H/32). Each wave: 16x16 h-tile, 4 gate accumulators.
__global__ __launch_bounds__(256)
void lstm_cell_wmma(const _Float16* __restrict__ h_prev,   // (B,H) f16
                    _Float16*       __restrict__ h_next,   // (B,H) f16
                    float*          __restrict__ c_state,  // (B,H) f32
                    const _Float16* __restrict__ whh,      // (4H,H) f16 row-major
                    const float*    __restrict__ xtab,     // (V,4H) x@Wih.T + bih + bhh
                    const int*      __restrict__ tok,      // token base (already + t)
                    int tokStride)
{
    __shared__ _Float16 bsm[2][4 * 32 * BSTRIDE];   // 2 x 18432B

    const int tid   = threadIdx.x;
    const int lane  = tid & 31;
    const int wave  = tid >> 5;
    const int waveM = wave & 3, waveN = wave >> 2;
    const int m0    = blockIdx.x * 64 + waveM * 16;
    const int n0b   = blockIdx.y * 32;
    const int n0    = n0b + waveN * 16;
    const int l16   = lane & 15, lh = lane >> 4;

    const size_t rowA = (size_t)(m0 + l16) * H_ + (lh ? 8 : 0);
    const int    lcol = waveN * 16 + l16;           // column within block tile
    const int    lkof = lh * 8;                     // K offset within fragment

    v8f acc[4] = {};

    // prologue: stage slice 0
    stage_b(whh, n0b, 0, bsm[0], tid);

    for (int kk = 0; kk < H_; kk += KSLICE) {
        const int cur = (kk >> 6) & 1;
        wait_async_lds();
        __syncthreads();                            // slice `cur` ready; also
                                                    // guarantees prior slice's
                                                    // readers finished before
                                                    // its buffer is restaged
        if (kk + KSLICE < H_)
            stage_b(whh, n0b, kk + KSLICE, bsm[cur ^ 1], tid);

        // issue both A fragments for this slice up front
        v16h a0 = load_a_frag(h_prev + rowA + kk);
        v16h a1 = load_a_frag(h_prev + rowA + kk + 32);

#pragma unroll
        for (int q = 0; q < 4; ++q) {
            const _Float16* lp = &bsm[cur][(q * 32 + lcol) * BSTRIDE + lkof];
            v16h b = load_b_frag16(lp);             // ds_load_b128 x2
            acc[q] = __builtin_amdgcn_wmma_f32_16x16x32_f16(
                false, a0, false, b, (short)0, acc[q], false, false);
        }
#pragma unroll
        for (int q = 0; q < 4; ++q) {
            const _Float16* lp = &bsm[cur][(q * 32 + lcol) * BSTRIDE + 32 + lkof];
            v16h b = load_b_frag16(lp);
            acc[q] = __builtin_amdgcn_wmma_f32_16x16x32_f16(
                false, a1, false, b, (short)0, acc[q], false, false);
        }
        // no trailing barrier: next iteration's top barrier orders restage
    }

    // epilogue: lane covers rows m0 + lh*8 + v (v=0..7), column n0 + l16.
    const int col = n0 + l16;
#pragma unroll
    for (int v = 0; v < 8; ++v) {
        const int row = m0 + lh * 8 + v;
        const int t   = tok[(size_t)row * tokStride];
        const float* xr = xtab + (size_t)t * G_ + col;
        float gi = acc[0][v] + xr[0 * H_];
        float gf = acc[1][v] + xr[1 * H_];
        float gg = acc[2][v] + xr[2 * H_];
        float go = acc[3][v] + xr[3 * H_];
        const size_t idx = (size_t)row * H_ + col;
        float cn = sigf(gf) * c_state[idx] + sigf(gi) * tanhf(gg);
        float hn = sigf(go) * tanhf(cn);
        c_state[idx] = cn;
        h_next[idx]  = (_Float16)hn;
    }
}

// ------------------- FC: logits[:, t_out, :] = h @ fc_w.T + fc_b -----------
// grid = (B/64, VO/32), same 8-wave tiling; operands direct from global (L2-hot).
__global__ __launch_bounds__(256)
void fc_wmma(const _Float16* __restrict__ h,     // (B,H) f16
             const _Float16* __restrict__ w,     // (VO,H) f16 row-major
             const float*    __restrict__ bias,  // (VO)
             float*          __restrict__ out,   // (B,T,VO)
             int t_out)
{
    const int lane  = threadIdx.x & 31;
    const int wave  = threadIdx.x >> 5;
    const int waveM = wave & 3, waveN = wave >> 2;
    const int m0    = blockIdx.x * 64 + waveM * 16;
    const int n0    = blockIdx.y * 32 + waveN * 16;
    const int l16   = lane & 15, lh = lane >> 4;

    const size_t rowA = (size_t)(m0 + l16) * H_ + (lh ? 8 : 0);
    const size_t colB = (size_t)(n0 + l16) * H_ + (size_t)lh * 8;

    v8f acc = {};
    v16h a_cur = load_a_frag(h + rowA);
    v16h b_cur = load_b_frag16(w + colB);
    for (int kk = 0; kk < H_; kk += 32) {
        v16h a_nxt = a_cur, b_nxt = b_cur;
        if (kk + 32 < H_) {
            a_nxt = load_a_frag(h + rowA + kk + 32);
            b_nxt = load_b_frag16(w + colB + kk + 32);
        }
        acc = __builtin_amdgcn_wmma_f32_16x16x32_f16(
            false, a_cur, false, b_cur, (short)0, acc, false, false);
        a_cur = a_nxt;
        b_cur = b_nxt;
    }

    const int col = n0 + l16;
    const float bn = bias[col];
#pragma unroll
    for (int v = 0; v < 8; ++v) {
        const int row = m0 + lh * 8 + v;
        out[(size_t)row * (T_ * VO_) + (size_t)t_out * VO_ + col] = acc[v] + bn;
    }
}

// ------------------- small prep kernels ------------------------------------
__global__ void build_table(const float* __restrict__ emb,
                            const float* __restrict__ wih,
                            const float* __restrict__ bih,
                            const float* __restrict__ bhh,
                            float* __restrict__ tab, int V)
{
    int tid = blockIdx.x * blockDim.x + threadIdx.x;
    if (tid >= V * G_) return;
    int v = tid / G_, n = tid % G_;
    float s = bih[n] + bhh[n];
    const float* e = emb + (size_t)v * E_;
    const float* w = wih + (size_t)n * E_;
#pragma unroll 4
    for (int k = 0; k < E_; ++k) s = fmaf(e[k], w[k], s);
    tab[tid] = s;
}

__global__ void f32_to_f16(const float* __restrict__ in, _Float16* __restrict__ out, int n) {
    int i = blockIdx.x * blockDim.x + threadIdx.x;
    if (i < n) out[i] = (_Float16)in[i];
}

__global__ void zero_f16(_Float16* p, int n) {
    int i = blockIdx.x * blockDim.x + threadIdx.x;
    if (i < n) p[i] = (_Float16)0.0f;
}

__global__ void zero_f32(float* p, int n) {
    int i = blockIdx.x * blockDim.x + threadIdx.x;
    if (i < n) p[i] = 0.0f;
}

__global__ void zero_out_col0(float* __restrict__ out) {
    int i = blockIdx.x * blockDim.x + threadIdx.x;
    if (i < B_ * VO_) {
        int b = i / VO_, n = i % VO_;
        out[(size_t)b * (T_ * VO_) + n] = 0.0f;
    }
}

// ---------------------------------------------------------------------------
extern "C" void kernel_launch(void* const* d_in, const int* in_sizes, int n_in,
                              void* d_out, int out_size, void* d_ws, size_t ws_size,
                              hipStream_t stream)
{
    const int*   src     = (const int*)  d_in[0];
    const int*   target  = (const int*)  d_in[1];
    const float* emb     = (const float*)d_in[2];
    const float* dec_emb = (const float*)d_in[3];
    const float* enc_wih = (const float*)d_in[4];
    const float* enc_whh = (const float*)d_in[5];
    const float* enc_bih = (const float*)d_in[6];
    const float* enc_bhh = (const float*)d_in[7];
    const float* dec_wih = (const float*)d_in[8];
    const float* dec_whh = (const float*)d_in[9];
    const float* dec_bih = (const float*)d_in[10];
    const float* dec_bhh = (const float*)d_in[11];
    const float* fc_w    = (const float*)d_in[12];
    const float* fc_b    = (const float*)d_in[13];
    float* out = (float*)d_out;

    // workspace carve-out (~10.6 MB total)
    char* wsp = (char*)d_ws;
    auto carve = [&](size_t bytes) {
        char* p = wsp;
        wsp += (bytes + 255) & ~(size_t)255;
        return (void*)p;
    };
    float*    enc_tab   = (float*)   carve((size_t)VI_ * G_ * 4);
    float*    dec_tab   = (float*)   carve((size_t)VO_ * G_ * 4);
    _Float16* enc_whh16 = (_Float16*)carve((size_t)G_ * H_ * 2);
    _Float16* dec_whh16 = (_Float16*)carve((size_t)G_ * H_ * 2);
    _Float16* fcw16     = (_Float16*)carve((size_t)VO_ * H_ * 2);
    float*    c_state   = (float*)   carve((size_t)B_ * H_ * 4);
    _Float16* hA        = (_Float16*)carve((size_t)B_ * H_ * 2);
    _Float16* hB        = (_Float16*)carve((size_t)B_ * H_ * 2);

    const int thr = 256;

    // weight conversion + tables + init
    f32_to_f16<<<(G_ * H_ + thr - 1) / thr, thr, 0, stream>>>(enc_whh, enc_whh16, G_ * H_);
    f32_to_f16<<<(G_ * H_ + thr - 1) / thr, thr, 0, stream>>>(dec_whh, dec_whh16, G_ * H_);
    f32_to_f16<<<(VO_ * H_ + thr - 1) / thr, thr, 0, stream>>>(fc_w, fcw16, VO_ * H_);
    build_table<<<(VI_ * G_ + thr - 1) / thr, thr, 0, stream>>>(emb, enc_wih, enc_bih, enc_bhh, enc_tab, VI_);
    build_table<<<(VO_ * G_ + thr - 1) / thr, thr, 0, stream>>>(dec_emb, dec_wih, dec_bih, dec_bhh, dec_tab, VO_);
    zero_f16<<<(B_ * H_ + thr - 1) / thr, thr, 0, stream>>>(hA, B_ * H_);
    zero_f32<<<(B_ * H_ + thr - 1) / thr, thr, 0, stream>>>(c_state, B_ * H_);
    zero_out_col0<<<(B_ * VO_ + thr - 1) / thr, thr, 0, stream>>>(out);

    dim3 cellGrid(B_ / 64, H_ / 32);   // (16,16)
    dim3 fcGrid(B_ / 64, VO_ / 32);    // (16,4)

    _Float16* hp = hA;
    _Float16* hn = hB;

    // encoder: 128 steps
    for (int t = 0; t < S_; ++t) {
        lstm_cell_wmma<<<cellGrid, 256, 0, stream>>>(hp, hn, c_state,
                                                     enc_whh16, enc_tab, src + t, S_);
        _Float16* tmp = hp; hp = hn; hn = tmp;
    }

    // decoder: 127 steps, FC per step into out[:, t+1, :]
    for (int t = 0; t < T_ - 1; ++t) {
        lstm_cell_wmma<<<cellGrid, 256, 0, stream>>>(hp, hn, c_state,
                                                     dec_whh16, dec_tab, target + t, T_);
        _Float16* tmp = hp; hp = hn; hn = tmp;   // hp now = fresh h_t
        fc_wmma<<<fcGrid, 256, 0, stream>>>(hp, fcw16, fc_b, out, t + 1);
    }
}
